// Impulse_72739566125608
// MI455X (gfx1250) — matrix operand order repaired and checked by reading, not compiled, not run on hardware
//
#include <hip/hip_runtime.h>

// ---------------------------------------------------------------------------
// Impulse-response convolution + level mix for MI455X (gfx1250, wave32).
//
//   conv[b,t]  = sum_j x[b,t+j] * h[j]                 (zero pad past T)
//   out[i,b,t] = conv[b,t]*lvl[i] + x[b,t]*(1-lvl[i])  out: [8,8,T]
//
// Compute-bound (1.13e11 FLOP vs ~92MB HBM traffic) => f16 WMMA, f32 accum.
// Implicit-GEMM Toeplitz formulation:
//   A[m,kk] = x[t0 + 16m + kk_g]     (16x32 f16 fragment, global loads)
//   B[kk,n] = h[kk_g - n]            (pre-swizzled impulse, TDM -> LDS, shared
//                                     by all 8 waves of the workgroup)
//   D[m,n]  = conv[t0 + 16m + n]
// ---------------------------------------------------------------------------

typedef __attribute__((ext_vector_type(16))) _Float16     v16h;
typedef __attribute__((ext_vector_type(8)))  _Float16     v8h;
typedef __attribute__((ext_vector_type(8)))  float        v8f;
typedef __attribute__((ext_vector_type(4)))  unsigned int v4u;
typedef __attribute__((ext_vector_type(8)))  int          v8i;
typedef __attribute__((ext_vector_type(4)))  int          v4i;

#define BS        8
#define T_LEN     320000
#define KIMP      22050
#define CH        16                      // kb-blocks per LDS chunk
#define NCH       44                      // chunks
#define KB_BLOCKS (CH * NCH)              // 704 padded kb blocks (kk_g < 22528)
#define KPAD      (KB_BLOCKS * 32)        // 22528
#define TP        (T_LEN + KPAD + 256)    // padded x length: 342784 (mult of 8)
#define NTILES    (T_LEN / 256)           // 1250 (exact)
#define CHUNK_HALVES (CH * 512)           // 8192 halves = 16KB per chunk
#define CHUNK_DW  (CH * 256)              // 4096 dwords per chunk

// --- pack x (f32) -> zero-padded f16 copy --------------------------------
__global__ void pack_x_f16(const float* __restrict__ x, _Float16* __restrict__ xp) {
    long i = (long)blockIdx.x * blockDim.x + threadIdx.x;
    if (i >= (long)BS * TP) return;
    int b = (int)(i / TP);
    int t = (int)(i % TP);
    float v = (t < T_LEN) ? x[(long)b * T_LEN + t] : 0.0f;
    xp[i] = (_Float16)v;
}

// --- build swizzled Toeplitz impulse: hsw[kb][lane][e] -------------------
// WMMA B layout (32x16 f16, wave32): lane L -> column n = L&15,
// halves e=0..7 -> kk = 8*(L>>4) + e, e=8..15 -> kk = 16 + 8*(L>>4) + (e-8)
__global__ void build_hsw(const float* __restrict__ h, _Float16* __restrict__ hsw) {
    int i = blockIdx.x * blockDim.x + threadIdx.x;   // = kb*512 + lane*16 + e
    if (i >= KB_BLOCKS * 512) return;
    int e    = i & 15;
    int lane = (i >> 4) & 31;
    int kb   = i >> 9;
    int n    = lane & 15;
    int kk   = ((lane >> 4) << 3) + (e & 7) + ((e >> 3) << 4);
    int idx  = kb * 32 + kk - n;                     // h index, 0 outside [0,KIMP)
    float v  = (idx >= 0 && idx < KIMP) ? h[idx] : 0.0f;
    hsw[i] = (_Float16)v;
}

// --- TDM: async 1-D copy of one 16KB impulse chunk into LDS ---------------
// D# per CDNA5 ISA ch.8: group0 {count=1, lds_addr, global_addr, type=2},
// group1 {data_size=4B, tensor_dim0=tile_dim0=4096 dwords, dims1+ unused}.
__device__ __forceinline__ void tdm_load_chunk(const _Float16* gsrc, unsigned lds_off) {
    unsigned long long ga = (unsigned long long)gsrc;
    v4u g0;
    g0.x = 1u;                                        // count=1, user descriptor
    g0.y = lds_off;                                   // lds_addr (bytes)
    g0.z = (unsigned)(ga & 0xFFFFFFFFu);              // global_addr[31:0]
    g0.w = (unsigned)((ga >> 32) & 0x01FFFFFFu)       // global_addr[56:32]
         | 0x80000000u;                               // type=2 ("image")
    v8i g1;
    g1[0] = (int)(2u << 16);                          // data_size = 4 bytes
    g1[1] = (int)((CHUNK_DW & 0xFFFF) << 16);         // tensor_dim0 lo16
    g1[2] = (int)(CHUNK_DW >> 16);                    // tensor_dim0 hi16; dim1=0
    g1[3] = (int)((CHUNK_DW & 0xFFFF) << 16);         // tile_dim0 = 4096 dwords
    g1[4] = 0;                                        // tile_dim1/2 unused
    g1[5] = (int)CHUNK_DW;                            // dim0 stride (single row)
    g1[6] = 0;
    g1[7] = 0;
    v4i gz = {0, 0, 0, 0};
#if __clang_major__ >= 23
    v8i gz8 = {0, 0, 0, 0, 0, 0, 0, 0};
    __builtin_amdgcn_tensor_load_to_lds(g0, g1, gz, gz, gz8, 0);
#else
    __builtin_amdgcn_tensor_load_to_lds(g0, g1, gz, gz, 0);
#endif
}

// --- main conv + broadcast-mix kernel ------------------------------------
// grid.x = NTILES (t0 = blockIdx.x*256), 256 threads = 8 waves, wave w -> b=w
__global__ __launch_bounds__(256) void conv_wmma_mix(
    const _Float16* __restrict__ xp,    // [BS][TP] padded f16 x
    const _Float16* __restrict__ hsw,   // [KB_BLOCKS][32][16] swizzled impulse
    const float*    __restrict__ x32,   // [BS][T] original x (dry path, f32)
    const float*    __restrict__ level, // [BS]
    float*          __restrict__ out)   // [BS][BS][T]
{
    __shared__ __align__(128) _Float16 bsm[2][CHUNK_HALVES];   // 32KB double buffer

    const int lane = threadIdx.x & 31;
    const int b    = threadIdx.x >> 5;       // wave id == batch
    const int t0   = blockIdx.x << 8;        // 256 outputs per tile
    const bool w0  = (threadIdx.x < 32);     // TDM-issuing wave

    // A fragment per-lane addressing (ISA 16-bit A 16x32 layout):
    //  lane L: m = L&15, c = L>>4; halves 0..7 -> kk=8c+e, 8..15 -> kk=8c+16+(e-8)
    const int m = lane & 15;
    const int c = lane >> 4;
    const _Float16* abase = xp + (long)b * TP + t0 + 16 * m + 8 * c;
    const _Float16* bsl0  = &bsm[0][lane * 16];
    const _Float16* bsl1  = &bsm[1][lane * 16];

    // Prologue: stream chunk 0 into buffer 0.
    if (w0) tdm_load_chunk(hsw, (unsigned)(unsigned long long)&bsm[0][0]);

    v8f acc0 = {}; v8f acc1 = {};   // two accumulators break the WMMA RAW chain

    for (int ch = 0; ch < NCH; ++ch) {
        __builtin_amdgcn_s_wait_tensorcnt(0);   // chunk ch landed (cnt 0 elsewhere)
        __syncthreads();                        // publish LDS chunk to all waves
        if (w0 && (ch + 1 < NCH)) {
            tdm_load_chunk(hsw + (long)(ch + 1) * CHUNK_HALVES,
                           (unsigned)(unsigned long long)&bsm[(ch + 1) & 1][0]);
        }

        const _Float16* bs = (ch & 1) ? bsl1 : bsl0;
        #pragma unroll
        for (int u = 0; u < CH; u += 2) {
            v8h a0 = *(const v8h*)(abase);
            v8h a1 = *(const v8h*)(abase + 16);
            v8h a2 = *(const v8h*)(abase + 32);
            v8h a3 = *(const v8h*)(abase + 48);
            v16h av0 = __builtin_shufflevector(a0, a1, 0,1,2,3,4,5,6,7,8,9,10,11,12,13,14,15);
            v16h av1 = __builtin_shufflevector(a2, a3, 0,1,2,3,4,5,6,7,8,9,10,11,12,13,14,15);
            v16h bv0 = *(const v16h*)(bs + (long)u * 512);
            v16h bv1 = *(const v16h*)(bs + (long)(u + 1) * 512);
            acc0 = __builtin_amdgcn_wmma_f32_16x16x32_f16(false, av0, false, bv0,
                                                          (short)0, acc0, false, false);
            acc1 = __builtin_amdgcn_wmma_f32_16x16x32_f16(false, av1, false, bv1,
                                                          (short)0, acc1, false, false);
            abase += 64;
        }
        __syncthreads();                        // all reads done before overwrite
    }

    // Merge accumulators: D[m,n] = conv[t0 + 16m + n]
    v8f acc;
    #pragma unroll
    for (int r = 0; r < 8; ++r) acc[r] = acc0[r] + acc1[r];

    // Epilogue: torch-style broadcast mix -> out[i][b][t]
    float lev[BS], oml[BS];
    #pragma unroll
    for (int i = 0; i < BS; ++i) { lev[i] = level[i]; oml[i] = 1.0f - lev[i]; }

    const int ncol    = lane & 15;           // D column
    const int mrow_hi = (lane >> 4) * 8;     // VGPR r: M = r + 8*(lane>>4)
    #pragma unroll
    for (int r = 0; r < 8; ++r) {
        const int  t  = t0 + 16 * (r + mrow_hi) + ncol;   // always < T
        const float cv = acc[r];
        const float xv = x32[(long)b * T_LEN + t];
        #pragma unroll
        for (int i = 0; i < BS; ++i) {
            out[((long)(i * BS + b)) * T_LEN + t] = cv * lev[i] + xv * oml[i];
        }
    }
}

extern "C" void kernel_launch(void* const* d_in, const int* in_sizes, int n_in,
                              void* d_out, int out_size, void* d_ws, size_t ws_size,
                              hipStream_t stream) {
    const float* x   = (const float*)d_in[0];   // [8, 320000]
    const float* h   = (const float*)d_in[1];   // [22050]
    const float* lvl = (const float*)d_in[2];   // [8]
    float* out = (float*)d_out;                 // [8, 8, 320000]

    // Workspace: f16 padded x copy (5.48MB), then swizzled impulse (0.72MB).
    _Float16* xp  = (_Float16*)d_ws;
    _Float16* hsw = xp + (size_t)BS * TP;       // 32B-aligned offset

    {
        long n = (long)BS * TP;
        pack_x_f16<<<(int)((n + 255) / 256), 256, 0, stream>>>(x, xp);
    }
    {
        int n = KB_BLOCKS * 512;
        build_hsw<<<(n + 255) / 256, 256, 0, stream>>>(h, hsw);
    }
    conv_wmma_mix<<<NTILES, 256, 0, stream>>>(xp, hsw, x, lvl, out);
}
